// KOMO_IK_39195871543737
// MI455X (gfx1250) — matrix operand order repaired and checked by reading, not compiled, not run on hardware
//
#include <hip/hip_runtime.h>
#include <math.h>

#define HD __device__ __forceinline__

typedef __attribute__((ext_vector_type(16))) __bf16 v16bf;
typedef __attribute__((ext_vector_type(8)))  float  v8f;

union Frag { v16bf v; uint4 q[2]; };

#define HDIM   512
#define TM     32          // batch rows per block (two 16-row WMMA M-tiles)
#define NITER  8
#define LSSTEP 12
#define NUV    10.0f

// ---------- bf16 helpers (round-to-nearest-even) ----------
HD unsigned short f2bf(float f) {
    unsigned int u = __float_as_uint(f);
    unsigned int r = (u + 0x7FFFu + ((u >> 16) & 1u)) >> 16;
    return (unsigned short)r;
}
HD float bf2f(unsigned short h) {
    return __uint_as_float(((unsigned int)h) << 16);
}

// ---------- quaternion scalar helpers ----------
HD void qmul(const float* a, const float* b, float* o) {
    o[0] = a[0]*b[0] - a[1]*b[1] - a[2]*b[2] - a[3]*b[3];
    o[1] = a[0]*b[1] + a[1]*b[0] + a[2]*b[3] - a[3]*b[2];
    o[2] = a[0]*b[2] - a[1]*b[3] + a[2]*b[0] + a[3]*b[1];
    o[3] = a[0]*b[3] + a[1]*b[2] - a[2]*b[1] + a[3]*b[0];
}
HD void aa2q(const float* aa, float* q) {
    float sq = aa[0]*aa[0] + aa[1]*aa[1] + aa[2]*aa[2];
    bool small = sq < 1e-12f;
    float ang  = small ? 0.0f : sqrtf(sq);
    float half = 0.5f * ang;
    float s    = small ? (0.5f - sq * (1.0f/48.0f)) : (sinf(half) / ang);
    q[0] = cosf(half);
    q[1] = aa[0]*s; q[2] = aa[1]*s; q[3] = aa[2]*s;
}
HD void q2aa(const float* q, float* aa) {
    float sq = q[1]*q[1] + q[2]*q[2] + q[3]*q[3];
    bool small = sq < 1e-12f;
    float norms = small ? 0.0f : sqrtf(sq);
    float half  = atan2f(norms, q[0]);
    float ang   = 2.0f * half;
    float s     = small ? (0.5f - ang*ang * (1.0f/48.0f)) : (sinf(half) / ang);
    float inv   = 1.0f / s;
    aa[0] = q[1]*inv; aa[1] = q[2]*inv; aa[2] = q[3]*inv;
}
HD void apply_delta(const float* x, const float* dx, float* o) {
    o[0] = x[0] + dx[0]; o[1] = x[1] + dx[1]; o[2] = x[2] + dx[2];
    float dq[4]; aa2q(dx + 3, dq);
    qmul(x + 3, dq, o + 3);
}
HD void phi_fn(const float* xt, const float* x0, float* phi) {
    phi[0] = xt[0]-x0[0]; phi[1] = xt[1]-x0[1]; phi[2] = xt[2]-x0[2];
    float qi[4] = { x0[3], -x0[4], -x0[5], -x0[6] };
    float qd[4]; qmul(qi, xt + 3, qd);
    float aa[3]; q2aa(qd, aa);
    phi[3] = 0.1f*aa[0]; phi[4] = 0.1f*aa[1]; phi[5] = 0.1f*aa[2];
}
HD void solve6(float A[6][6], const float* b, float* x) {
    float M[6][7];
    for (int i = 0; i < 6; ++i) { for (int j = 0; j < 6; ++j) M[i][j] = A[i][j]; M[i][6] = b[i]; }
    for (int c = 0; c < 6; ++c) {
        int p = c; float mx = fabsf(M[c][c]);
        for (int r = c + 1; r < 6; ++r) { float v = fabsf(M[r][c]); if (v > mx) { mx = v; p = r; } }
        if (p != c) for (int j = c; j < 7; ++j) { float t = M[c][j]; M[c][j] = M[p][j]; M[p][j] = t; }
        float inv = 1.0f / M[c][c];
        for (int j = c; j < 7; ++j) M[c][j] *= inv;
        for (int r = 0; r < 6; ++r) {
            if (r == c) continue;
            float f = M[r][c];
            for (int j = c; j < 7; ++j) M[r][j] -= f * M[c][j];
        }
    }
    for (int i = 0; i < 6; ++i) x[i] = M[i][6];
}

// ---------- WMMA GEMM: (32 x 512) x (512 x 512) in bf16, f32 accumulate ----------
// inbuf: LDS row-major bf16 [32][512]; pack: pre-swizzled B tiles in global.
// Wave w owns N-tiles w*4 .. w*4+3 for BOTH 16-row M-tiles: each B fragment
// feeds 2 WMMAs -> 2x arithmetic intensity vs a single M-tile. K-loop rolled
// (unroll 1) so only one K-step of fragments is live -> no VGPR spill.
HD void gemm32(const unsigned short* inbuf, const unsigned short* __restrict__ pack,
               const float* __restrict__ bias, unsigned short* outbuf, bool act, int tid)
{
    const int lane = tid & 31;
    const int w    = tid >> 5;
    const int row  = lane & 15;
    const int hi   = lane >> 4;
    v8f acc[8] = {};
    #pragma unroll 1
    for (int kt = 0; kt < 16; ++kt) {
        Frag a0, a1;
        const unsigned short* ap0 = inbuf + row * HDIM + kt * 32 + hi * 8;
        const unsigned short* ap1 = ap0 + 16 * HDIM;
        a0.q[0] = *(const uint4*)(ap0);
        a0.q[1] = *(const uint4*)(ap0 + 16);
        a1.q[0] = *(const uint4*)(ap1);
        a1.q[1] = *(const uint4*)(ap1 + 16);
        const unsigned short* bp = pack + ((size_t)(kt * 32 + w * 4) * 512) + lane * 16;
        __builtin_prefetch(bp + 32 * 512, 0, 3);   // next K-step tile
        Frag fb[4];
        #pragma unroll
        for (int j = 0; j < 4; ++j) {
            const unsigned short* bj = bp + (size_t)j * 512;
            fb[j].q[0] = *(const uint4*)(bj);
            fb[j].q[1] = *(const uint4*)(bj + 8);
        }
        #pragma unroll
        for (int j = 0; j < 4; ++j) {
            acc[j]     = __builtin_amdgcn_wmma_f32_16x16x32_bf16(
                false, a0.v, false, fb[j].v, (short)0, acc[j],     false, false);
            acc[4 + j] = __builtin_amdgcn_wmma_f32_16x16x32_bf16(
                false, a1.v, false, fb[j].v, (short)0, acc[4 + j], false, false);
        }
    }
    #pragma unroll
    for (int j = 0; j < 4; ++j) {
        int colb = (w * 4 + j) * 16 + row;
        float bb = bias[colb];
        #pragma unroll
        for (int r = 0; r < 8; ++r) {
            int m = hi ? (r + 8) : r;
            float v0 = acc[j][r] + bb;
            float v1 = acc[4 + j][r] + bb;
            if (act) { v0 = tanhf(v0); v1 = tanhf(v1); }
            outbuf[m * HDIM + colb]        = f2bf(v0);
            outbuf[(m + 16) * HDIM + colb] = f2bf(v1);
        }
    }
}

// ---------- full MLP forward for the 32-row tile ----------
HD void mlp_forward(const float* xin, const float* __restrict__ W1, const float* __restrict__ b1,
                    const float* __restrict__ b2, const float* __restrict__ W3,
                    const float* __restrict__ b3, const unsigned short* __restrict__ packF,
                    unsigned short* h1buf, unsigned short* h2buf, float* hout, int tid)
{
    // layer1: 7 -> 512, scalar
    for (int idx = tid; idx < TM * HDIM; idx += 256) {
        int r = idx >> 9, c = idx & 511;
        float s = b1[c];
        #pragma unroll
        for (int i = 0; i < 7; ++i) s += xin[r * 7 + i] * W1[i * HDIM + c];
        h1buf[idx] = f2bf(tanhf(s));
    }
    __syncthreads();
    // layer2: 512 -> 512, WMMA
    gemm32(h1buf, packF, b2, h2buf, true, tid);
    __syncthreads();
    // layer3: 512 -> 1, LDS reduction (32 rows x 8 segments of 64 cols)
    if (tid < TM) hout[tid] = 0.0f;
    __syncthreads();
    {
        int r = tid >> 3, seg = tid & 7;
        float s = 0.0f;
        int c0 = seg * 64;
        for (int c = c0; c < c0 + 64; ++c) s += bf2f(h2buf[r * HDIM + c]) * W3[c];
        atomicAdd(&hout[r], s);
    }
    __syncthreads();
    if (tid < TM) hout[tid] += b3[0];
    __syncthreads();
}

// ---------- MLP backward: grad of h wrt 7-dim input, per row ----------
HD void mlp_backward(const unsigned short* h1buf, const unsigned short* h2buf,
                     unsigned short* vbuf, const unsigned short* __restrict__ packB,
                     const float* __restrict__ W1, const float* __restrict__ W3,
                     float* gxbuf, int tid)
{
    // v2 = (1 - h2^2) * W3
    for (int idx = tid; idx < TM * HDIM; idx += 256) {
        int c = idx & 511;
        float h2 = bf2f(h2buf[idx]);
        vbuf[idx] = f2bf((1.0f - h2 * h2) * W3[c]);
    }
    if (tid < TM * 7) gxbuf[tid] = 0.0f;
    __syncthreads();
    // v1 = v2 @ W2^T via WMMA, then u1 = (1-h1^2)*v1, gx = u1 @ W1^T
    const int lane = tid & 31;
    const int w    = tid >> 5;
    const int row  = lane & 15;
    const int hi   = lane >> 4;
    v8f acc[8] = {};
    #pragma unroll 1
    for (int kt = 0; kt < 16; ++kt) {
        Frag a0, a1;
        const unsigned short* ap0 = vbuf + row * HDIM + kt * 32 + hi * 8;
        const unsigned short* ap1 = ap0 + 16 * HDIM;
        a0.q[0] = *(const uint4*)(ap0);
        a0.q[1] = *(const uint4*)(ap0 + 16);
        a1.q[0] = *(const uint4*)(ap1);
        a1.q[1] = *(const uint4*)(ap1 + 16);
        const unsigned short* bp = packB + ((size_t)(kt * 32 + w * 4) * 512) + lane * 16;
        __builtin_prefetch(bp + 32 * 512, 0, 3);
        Frag fb[4];
        #pragma unroll
        for (int j = 0; j < 4; ++j) {
            const unsigned short* bj = bp + (size_t)j * 512;
            fb[j].q[0] = *(const uint4*)(bj);
            fb[j].q[1] = *(const uint4*)(bj + 8);
        }
        #pragma unroll
        for (int j = 0; j < 4; ++j) {
            acc[j]     = __builtin_amdgcn_wmma_f32_16x16x32_bf16(
                false, a0.v, false, fb[j].v, (short)0, acc[j],     false, false);
            acc[4 + j] = __builtin_amdgcn_wmma_f32_16x16x32_bf16(
                false, a1.v, false, fb[j].v, (short)0, acc[4 + j], false, false);
        }
    }
    #pragma unroll
    for (int j = 0; j < 4; ++j) {
        int colb = (w * 4 + j) * 16 + row;
        float w1c[7];
        #pragma unroll
        for (int c = 0; c < 7; ++c) w1c[c] = W1[c * HDIM + colb];
        #pragma unroll
        for (int r = 0; r < 8; ++r) {
            int m = hi ? (r + 8) : r;
            float h1a = bf2f(h1buf[m * HDIM + colb]);
            float h1b = bf2f(h1buf[(m + 16) * HDIM + colb]);
            float ua  = (1.0f - h1a * h1a) * acc[j][r];
            float ub  = (1.0f - h1b * h1b) * acc[4 + j][r];
            #pragma unroll
            for (int c = 0; c < 7; ++c) {
                atomicAdd(&gxbuf[m * 7 + c],        ua * w1c[c]);
                atomicAdd(&gxbuf[(m + 16) * 7 + c], ub * w1c[c]);
            }
        }
    }
    __syncthreads();
}

// ---------- weight pack: W2 -> bf16 B-fragment layout (fwd and transposed) ----------
__global__ __launch_bounds__(256) void pack_w2_kernel(const float* __restrict__ W2,
                                                      unsigned short* __restrict__ pF,
                                                      unsigned short* __restrict__ pB)
{
    int idx = blockIdx.x * 256 + threadIdx.x;       // 0 .. 512*512-1
    int tile   = idx >> 9;
    int within = idx & 511;
    int l = within >> 4, h = within & 15;
    int kt = tile >> 5, nt = tile & 31;
    int k = kt * 32 + ((l >> 4) << 3) + (h & 7) + ((h >> 3) << 4);   // lanes>=16 hold K+8 (table order)
    int n = nt * 16 + (l & 15);
    pF[idx] = f2bf(W2[k * HDIM + n]);               // B[k][n] = W2[k][n]
    pB[idx] = f2bf(W2[n * HDIM + k]);               // B[k][j] = W2[j][k]  (W2^T)
}

// ---------- main solver kernel: one block = 32 batch rows ----------
__global__ __launch_bounds__(256) void komo_ik_kernel(
    const float* __restrict__ x_in, const float* __restrict__ W1, const float* __restrict__ b1,
    const float* __restrict__ b2, const float* __restrict__ W3, const float* __restrict__ b3,
    const unsigned short* __restrict__ packF, const unsigned short* __restrict__ packB,
    float* __restrict__ out)
{
    __shared__ unsigned short h1buf[TM * HDIM];   // 32 KB
    __shared__ unsigned short h2buf[TM * HDIM];   // 32 KB
    __shared__ unsigned short vbuf [TM * HDIM];   // 32 KB
    __shared__ float xloc[TM * 7], x0loc[TM * 7], xtloc[TM * 7];
    __shared__ float hsum[TM];
    __shared__ float gxbuf[TM * 7];
    __shared__ float deltas[TM * 6];
    __shared__ float Lrow[TM], sloperow[TM], alpharow[TM], kapparow[TM];

    const int tid   = threadIdx.x;
    const int rbase = blockIdx.x * TM;

    for (int i = tid; i < TM * 7; i += 256) {
        float v = x_in[rbase * 7 + i];
        xloc[i] = v; x0loc[i] = v;
    }
    if (tid < TM) kapparow[tid] = 0.0f;
    __syncthreads();

    for (int iter = 0; iter < NITER; ++iter) {
        // forward at current xc (also supplies h(x_new) for the dual update)
        mlp_forward(xloc, W1, b1, b2, W3, b3, packF, h1buf, h2buf, hsum, tid);
        if (iter > 0) {
            if (tid < TM) kapparow[tid] += 2.0f * NUV * hsum[tid];
            __syncthreads();
        }
        // backward (uses h1/h2 from this forward)
        mlp_backward(h1buf, h2buf, vbuf, packB, W1, W3, gxbuf, tid);

        // per-row scalar Newton step (one full wave: 32 rows)
        if (tid < TM) {
            const float* xc = &xloc[tid * 7];
            const float* x0 = &x0loc[tid * 7];
            float h   = hsum[tid];
            float kap = kapparow[tid];
            float phi[6]; phi_fn(xc, x0, phi);

            // J_phi by central differences (cheap scalar quaternion math)
            float Jphi[6][6];
            const float eps = 1e-4f, ieps = 0.5f / 1e-4f;
            for (int i = 0; i < 6; ++i) {
                float dxv[6] = {0,0,0,0,0,0};
                float xp[7], xm[7], pp[6], pm[6];
                dxv[i] =  eps; apply_delta(xc, dxv, xp);
                dxv[i] = -eps; apply_delta(xc, dxv, xm);
                phi_fn(xp, x0, pp); phi_fn(xm, x0, pm);
                for (int k = 0; k < 6; ++k) Jphi[k][i] = (pp[k] - pm[k]) * ieps;
            }
            // J_h: position part direct; rotation via d q / d aa at 0 = 0.5 q (x) e_i
            float Jh[6];
            Jh[0] = gxbuf[tid*7+0]; Jh[1] = gxbuf[tid*7+1]; Jh[2] = gxbuf[tid*7+2];
            float gq[4] = { gxbuf[tid*7+3], gxbuf[tid*7+4], gxbuf[tid*7+5], gxbuf[tid*7+6] };
            for (int i = 0; i < 3; ++i) {
                float e[4] = {0,0,0,0}; e[1+i] = 0.5f;
                float qe[4]; qmul(xc + 3, e, qe);
                Jh[3+i] = gq[0]*qe[0] + gq[1]*qe[1] + gq[2]*qe[2] + gq[3]*qe[3];
            }
            float L = h*kap + NUV*h*h;
            for (int k = 0; k < 6; ++k) L += phi[k]*phi[k];
            float coef = 2.0f * NUV * h + kap;
            float Lx[6], Lxx[6][6];
            for (int i = 0; i < 6; ++i) {
                float s = coef * Jh[i];
                for (int k = 0; k < 6; ++k) s += 2.0f * Jphi[k][i] * phi[k];
                Lx[i] = s;
            }
            for (int i = 0; i < 6; ++i)
                for (int j = 0; j < 6; ++j) {
                    float s = 2.0f * NUV * Jh[i] * Jh[j];
                    for (int k = 0; k < 6; ++k) s += 2.0f * Jphi[k][i] * Jphi[k][j];
                    Lxx[i][j] = s;
                }
            float dsol[6]; solve6(Lxx, Lx, dsol);
            float slope = 0.0f;
            for (int i = 0; i < 6; ++i) {
                float d = -dsol[i];
                deltas[tid*6+i] = d;
                slope += Lx[i] * d;
            }
            Lrow[tid] = L; sloperow[tid] = slope; alpharow[tid] = 1.0f;
        }
        __syncthreads();

        // fixed 12-step backtracking line search, each trial re-runs the MLP
        for (int ls = 0; ls < LSSTEP; ++ls) {
            if (tid < TM) {
                float a = alpharow[tid];
                float dxv[6];
                for (int d = 0; d < 6; ++d) dxv[d] = a * deltas[tid*6+d];
                float xt[7]; apply_delta(&xloc[tid*7], dxv, xt);
                for (int i = 0; i < 7; ++i) xtloc[tid*7+i] = xt[i];
            }
            __syncthreads();
            mlp_forward(xtloc, W1, b1, b2, W3, b3, packF, h1buf, h2buf, hsum, tid);
            if (tid < TM) {
                float phit[6]; phi_fn(&xtloc[tid*7], &x0loc[tid*7], phit);
                float ht = hsum[tid];
                float ft = ht*kapparow[tid] + NUV*ht*ht;
                for (int k = 0; k < 6; ++k) ft += phit[k]*phit[k];
                if (ft > Lrow[tid] + 0.5f * alpharow[tid] * sloperow[tid])
                    alpharow[tid] *= 0.5f;
            }
            __syncthreads();
        }

        // accept step
        if (tid < TM) {
            float a = alpharow[tid];
            float dxv[6];
            for (int d = 0; d < 6; ++d) dxv[d] = a * deltas[tid*6+d];
            float xn[7]; apply_delta(&xloc[tid*7], dxv, xn);
            for (int i = 0; i < 7; ++i) xloc[tid*7+i] = xn[i];
        }
        __syncthreads();
    }

    if (tid < TM * 7) out[rbase * 7 + tid] = xloc[tid];
}

extern "C" void kernel_launch(void* const* d_in, const int* in_sizes, int n_in,
                              void* d_out, int out_size, void* d_ws, size_t ws_size,
                              hipStream_t stream)
{
    const float* x  = (const float*)d_in[0];   // (8192, 7)
    const float* W1 = (const float*)d_in[1];   // (7, 512)
    const float* b1 = (const float*)d_in[2];   // (512,)
    const float* W2 = (const float*)d_in[3];   // (512, 512)
    const float* b2 = (const float*)d_in[4];   // (512,)
    const float* W3 = (const float*)d_in[5];   // (512, 1)
    const float* b3 = (const float*)d_in[6];   // (1,)
    float* out = (float*)d_out;

    unsigned short* packF = (unsigned short*)d_ws;                 // 512 KB
    unsigned short* packB = packF + (size_t)HDIM * HDIM;           // 512 KB

    // pre-swizzle W2 / W2^T into WMMA B-fragment layout (bf16)
    pack_w2_kernel<<<(HDIM * HDIM) / 256, 256, 0, stream>>>(W2, packF, packB);

    // 8192 rows / 32 rows-per-block = 256 blocks, 256 threads (8 wave32s)
    komo_ik_kernel<<<8192 / TM, 256, 0, stream>>>(x, W1, b1, b2, W3, b3,
                                                  packF, packB, out);
}
